// GATLayer_71305047048239
// MI455X (gfx1250) — matrix operand order repaired and checked by reading, not compiled, not run on hardware
//
#include <hip/hip_runtime.h>

#define N_NODES 50000
#define N_EDGES 800000
#define IN_DIM  256
#define OUT_DIM 32
#define NUM_HEADS 8

typedef __attribute__((ext_vector_type(16))) __bf16 v16bf;
typedef __attribute__((ext_vector_type(8)))  float  v8f;

union Frag {
    uint4 q[2];
    v16bf v;
};

static __device__ __forceinline__ unsigned short f2bf(float f) {
    unsigned u = __float_as_uint(f);
    unsigned r = u + 0x7FFFu + ((u >> 16) & 1u);   // round-to-nearest-even
    return (unsigned short)(r >> 16);
}

static __device__ __forceinline__ float lrelu(float x, float slope) {
    return x > 0.0f ? x : slope * x;
}

// Ordered-uint encoding so unsigned atomicMax == float max (handles negatives).
static __device__ __forceinline__ unsigned ordEnc(float f) {
    unsigned u = __float_as_uint(f);
    return (u & 0x80000000u) ? ~u : (u | 0x80000000u);
}
static __device__ __forceinline__ float ordDec(unsigned e) {
    unsigned u = (e & 0x80000000u) ? (e & 0x7FFFFFFFu) : ~e;
    return __uint_as_float(u);
}

// ---------------- prep kernels ----------------

__global__ void k_f2bf(const float* __restrict__ src, unsigned short* __restrict__ dst, int n) {
    int i = blockIdx.x * blockDim.x + threadIdx.x;
    if (i < n) dst[i] = f2bf(src[i]);
}

// Swizzle a K x N = 256x256 f32 weight matrix into per-lane WMMA B-fragment order:
//   out[((ntile*8 + ktile)*32 + lane)*16 + (2*v + j)] = W[(ktile*32 + 16*(lane>>4) + 2*v + j)*256 + ntile*16 + (lane&15)]
__global__ void k_swizzleW(const float* __restrict__ W, unsigned short* __restrict__ out) {
    int o = blockIdx.x * blockDim.x + threadIdx.x;   // 65536 total
    int e    = o & 15;
    int lane = (o >> 4) & 31;
    int kt   = (o >> 9) & 7;
    int nt   = o >> 12;
    int v = e >> 1, j = e & 1;
    int n = nt * 16 + (lane & 15);
    int hh = lane >> 4;
    int k = kt * 32 + 16 * hh + 2 * v + j;
    out[o] = f2bf(W[k * IN_DIM + n]);
}

// ---------------- bf16 WMMA GEMM: C[M,256] = A[M,256] * B[256,256] (+bias) ----------------
// grid = M/16 blocks, 256 threads (8 waves). Wave w computes N-tiles 2w, 2w+1.

__global__ void k_gemm_bf16(const unsigned short* __restrict__ A,
                            const unsigned short* __restrict__ Bsw,
                            const float* __restrict__ bias,
                            float* __restrict__ Cf,
                            unsigned short* __restrict__ Cb) {
    const int lane = threadIdx.x & 31;
    const int wave = threadIdx.x >> 5;
    const int half = lane >> 4;
    const int l15  = lane & 15;
    const int row  = blockIdx.x * 16 + l15;

    const unsigned* arow = reinterpret_cast<const unsigned*>(A) + (size_t)row * (IN_DIM / 2);

    v8f acc0 = {};
    v8f acc1 = {};

    const int nt0 = wave * 2;
    const int nt1 = nt0 + 1;
    const uint4* bp0 = reinterpret_cast<const uint4*>(Bsw) + ((size_t)(nt0 * 8) * 32 + lane) * 2;
    const uint4* bp1 = reinterpret_cast<const uint4*>(Bsw) + ((size_t)(nt1 * 8) * 32 + lane) * 2;

#pragma unroll
    for (int kt = 0; kt < 8; ++kt) {
        Frag a;
        // A layout (matches swizzled B): VGPR0-7 -> K = 16*half + {0..15}
        a.q[0] = *reinterpret_cast<const uint4*>(arow + kt * 16 + 8 * half);
        a.q[1] = *reinterpret_cast<const uint4*>(arow + kt * 16 + 8 * half + 4);
        Frag b0, b1;
        b0.q[0] = bp0[0]; b0.q[1] = bp0[1];
        b1.q[0] = bp1[0]; b1.q[1] = bp1[1];
        bp0 += 64; bp1 += 64;   // advance one ktile (32 lanes * 32B)
        acc0 = __builtin_amdgcn_wmma_f32_16x16x32_bf16(false, a.v, false, b0.v,
                                                       (short)0, acc0, false, false);
        acc1 = __builtin_amdgcn_wmma_f32_16x16x32_bf16(false, a.v, false, b1.v,
                                                       (short)0, acc1, false, false);
    }

    // C layout: VGPR r, lanes 0-15 -> M = r, lanes 16-31 -> M = 8 + r; N = lane&15
    const int orow = blockIdx.x * 16 + 8 * half;
#pragma unroll
    for (int r = 0; r < 8; ++r) {
        int rr = orow + r;
        int c0 = nt0 * 16 + l15;
        int c1 = nt1 * 16 + l15;
        float v0 = acc0[r] + (bias ? bias[c0] : 0.0f);
        float v1 = acc1[r] + (bias ? bias[c1] : 0.0f);
        Cf[(size_t)rr * IN_DIM + c0] = v0;
        Cf[(size_t)rr * IN_DIM + c1] = v1;
        if (Cb) {
            Cb[(size_t)rr * IN_DIM + c0] = f2bf(v0);
            Cb[(size_t)rr * IN_DIM + c1] = f2bf(v1);
        }
    }
}

// ---------------- attention scalar projections ----------------

__global__ void k_elr(const float* __restrict__ feat,
                      const float* __restrict__ attn_l,
                      const float* __restrict__ attn_r,
                      float* __restrict__ el, float* __restrict__ er) {
    int t = blockIdx.x * blockDim.x + threadIdx.x;   // N*H
    if (t >= N_NODES * NUM_HEADS) return;
    int n = t >> 3, h = t & 7;
    const float* f = feat + (size_t)n * IN_DIM + h * OUT_DIM;
    const float* al = attn_l + h * OUT_DIM;
    const float* ar = attn_r + h * OUT_DIM;
    float sl = 0.0f, sr = 0.0f;
#pragma unroll
    for (int d = 0; d < OUT_DIM; ++d) {
        float fv = f[d];
        sl += fv * al[d];
        sr += fv * ar[d];
    }
    el[t] = sl;
    er[t] = sr;
}

// ---------------- edge softmax passes ----------------

__global__ void k_edge_max(const int* __restrict__ src, const int* __restrict__ dst,
                           const float* __restrict__ el, const float* __restrict__ er,
                           unsigned* __restrict__ menc) {
    int t = blockIdx.x * blockDim.x + threadIdx.x;   // E*H
    if (t >= N_EDGES * NUM_HEADS) return;
    int e = t >> 3, h = t & 7;
    int s = src[e], d = dst[e];
    float x = lrelu(el[s * NUM_HEADS + h] + er[d * NUM_HEADS + h], 0.2f);
    atomicMax(&menc[d * NUM_HEADS + h], ordEnc(x));
}

__global__ void k_edge_sum(const int* __restrict__ src, const int* __restrict__ dst,
                           const float* __restrict__ el, const float* __restrict__ er,
                           const unsigned* __restrict__ menc, float* __restrict__ ssum) {
    int t = blockIdx.x * blockDim.x + threadIdx.x;   // E*H
    if (t >= N_EDGES * NUM_HEADS) return;
    int e = t >> 3, h = t & 7;
    int s = src[e], d = dst[e];
    float x = lrelu(el[s * NUM_HEADS + h] + er[d * NUM_HEADS + h], 0.2f);
    float m = ordDec(menc[d * NUM_HEADS + h]);
    atomicAdd(&ssum[d * NUM_HEADS + h], __expf(x - m));
}

// One block per edge: 8 heads x 32 dims. Accumulates directly into out[N,256].
__global__ void k_edge_aggr(const int* __restrict__ src, const int* __restrict__ dst,
                            const float* __restrict__ el, const float* __restrict__ er,
                            const unsigned* __restrict__ menc, const float* __restrict__ ssum,
                            const float* __restrict__ feat, float* __restrict__ out) {
    int e = blockIdx.x;
    int h = threadIdx.x >> 5;
    int d = threadIdx.x & 31;
    int s = src[e], dd = dst[e];
    float x = lrelu(el[s * NUM_HEADS + h] + er[dd * NUM_HEADS + h], 0.2f);
    float m = ordDec(menc[dd * NUM_HEADS + h]);
    float a = __expf(x - m) / ssum[dd * NUM_HEADS + h];
    float msg = feat[(size_t)s * IN_DIM + h * OUT_DIM + d] * a;
    atomicAdd(&out[(size_t)dd * IN_DIM + h * OUT_DIM + d], msg);
}

// ---------------- epilogue: bias + leaky_relu(0.01) + residual ----------------

__global__ void k_finalize(float* __restrict__ out, const float* __restrict__ h1,
                           const float* __restrict__ gat_bias, int n) {
    int i = blockIdx.x * blockDim.x + threadIdx.x;
    if (i >= n) return;
    int col = i & (IN_DIM - 1);
    float v = lrelu(out[i] + gat_bias[col], 0.01f);
    out[i] = h1[i] + v;
}

// ---------------- host launch ----------------

extern "C" void kernel_launch(void* const* d_in, const int* in_sizes, int n_in,
                              void* d_out, int out_size, void* d_ws, size_t ws_size,
                              hipStream_t stream) {
    const float* h       = (const float*)d_in[0];
    const int*   src     = (const int*)  d_in[1];
    const int*   dst     = (const int*)  d_in[2];
    const float* W_lin   = (const float*)d_in[3];
    const float* b_lin   = (const float*)d_in[4];
    const float* W_gat   = (const float*)d_in[5];
    const float* attn_l  = (const float*)d_in[6];
    const float* attn_r  = (const float*)d_in[7];
    const float* gatbias = (const float*)d_in[8];
    float* out = (float*)d_out;

    const size_t NC = (size_t)N_NODES * IN_DIM;      // 12.8M elems
    char* ws = (char*)d_ws;
    size_t off = 0;
    auto alloc = [&](size_t bytes) { void* p = ws + off; off += (bytes + 255) & ~(size_t)255; return p; };

    unsigned short* hb      = (unsigned short*)alloc(NC * 2);                 // bf16 h
    float*          h1      = (float*)         alloc(NC * 4);                 // h @ W_lin + b
    unsigned short* h1b     = (unsigned short*)alloc(NC * 2);                 // bf16 h1
    float*          feat    = (float*)         alloc(NC * 4);                 // h1 @ W_gat
    unsigned short* WlinSw  = (unsigned short*)alloc((size_t)IN_DIM * IN_DIM * 2);
    unsigned short* WgatSw  = (unsigned short*)alloc((size_t)IN_DIM * IN_DIM * 2);
    float*          el      = (float*)   alloc((size_t)N_NODES * NUM_HEADS * 4);
    float*          er      = (float*)   alloc((size_t)N_NODES * NUM_HEADS * 4);
    unsigned*       menc    = (unsigned*)alloc((size_t)N_NODES * NUM_HEADS * 4);
    float*          ssum    = (float*)   alloc((size_t)N_NODES * NUM_HEADS * 4);

    // prep
    k_f2bf<<<(int)((NC + 255) / 256), 256, 0, stream>>>(h, hb, (int)NC);
    k_swizzleW<<<256, 256, 0, stream>>>(W_lin, WlinSw);
    k_swizzleW<<<256, 256, 0, stream>>>(W_gat, WgatSw);

    // GEMM1: h1 = h @ W_lin + b_lin (also bf16 copy for GEMM2)
    k_gemm_bf16<<<N_NODES / 16, 256, 0, stream>>>(hb, WlinSw, b_lin, h1, h1b);
    // GEMM2: feat = h1 @ W_gat
    k_gemm_bf16<<<N_NODES / 16, 256, 0, stream>>>(h1b, WgatSw, nullptr, feat, nullptr);

    // per-node attention scalars
    k_elr<<<(N_NODES * NUM_HEADS + 255) / 256, 256, 0, stream>>>(feat, attn_l, attn_r, el, er);

    // zero atomic targets (capturable async memsets)
    hipMemsetAsync(menc, 0, (size_t)N_NODES * NUM_HEADS * 4, stream);
    hipMemsetAsync(ssum, 0, (size_t)N_NODES * NUM_HEADS * 4, stream);
    hipMemsetAsync(out, 0, NC * 4, stream);

    // edge softmax + aggregation
    k_edge_max <<<(N_EDGES * NUM_HEADS + 255) / 256, 256, 0, stream>>>(src, dst, el, er, menc);
    k_edge_sum <<<(N_EDGES * NUM_HEADS + 255) / 256, 256, 0, stream>>>(src, dst, el, er, menc, ssum);
    k_edge_aggr<<<N_EDGES, 256, 0, stream>>>(src, dst, el, er, menc, ssum, feat, out);

    // bias + leaky_relu + residual
    k_finalize<<<(int)((NC + 255) / 256), 256, 0, stream>>>(out, h1, gatbias, (int)NC);
}
